// Attention_13185549599132
// MI455X (gfx1250) — compile-verified, matrix operand order checked by reading, not run on hardware
//
#include <hip/hip_runtime.h>
#include <hip/hip_bf16.h>

// ---------------------------------------------------------------------------
// MI455X (gfx1250) fused attention:
//   wcvt (f32->f16 transposed weights) -> LN -> QKV (WMMA, pipelined async-LDS)
//   -> V transpose -> flash attention (WMMA, pipelined async-LDS K/V)
//   -> out-proj (WMMA)
// wave32, v_wmma_f32_16x16x32_f16 fp32-accumulate,
// double-buffered GLOBAL_LOAD_ASYNC_TO_LDS_B128 + s_wait_asynccnt staging.
// ---------------------------------------------------------------------------

typedef __attribute__((ext_vector_type(16))) _Float16 v16h;
typedef __attribute__((ext_vector_type(8)))  float    v8f;

#define B_   2
#define N_   2048
#define D_   1024
#define H_   16
#define DH_  64
#define NQKV 3072
#define NEGV (-1e8f)

#define LSTR 40   // f16 stride, 32-wide tiles: 80B rows, 16B aligned, (20r+c)%64 distinct
#define KSTR 72   // f16 stride, 64-wide tiles: 144B rows, 16B aligned, (36r+c)%64 distinct

union Frag16 { v16h h; unsigned u[8]; };

// K-pair base for A/B f16 16x32 fragment layout (ISA 7.12.2):
// lanes 0-15: V0..3 -> K 0..7, V4..7 -> K 16..23 ; lanes 16-31: +8.
__device__ __forceinline__ int kpair(int v, int khalf) {
    return ((v < 4) ? (2 * v) : (2 * v + 8)) + khalf * 8;
}

__device__ __forceinline__ v8f wmma_f16(const Frag16& a, const Frag16& b, v8f c) {
    return __builtin_amdgcn_wmma_f32_16x16x32_f16(false, a.h, false, b.h,
                                                  (short)0, c, false, false);
}

__device__ __forceinline__ unsigned lds_off(const void* p) {
    // Flat LDS aperture: addr[31:0] is the LDS offset.
    return (unsigned)(unsigned long long)p;
}

__device__ __forceinline__ void async_b128(unsigned loff, unsigned long long gaddr) {
    asm volatile("global_load_async_to_lds_b128 %0, %1, off"
                 :: "v"(loff), "v"(gaddr) : "memory");
}

__device__ __forceinline__ void wait_async0() {
    asm volatile("s_wait_asynccnt 0" ::: "memory");
}

// ---------------------------------------------------------------------------
// Kernel 0: weight convert+transpose: w[K][Nn] f32 -> wt[Nn][K] f16.
// ---------------------------------------------------------------------------
__global__ void wcvt_kernel(const float* __restrict__ w, _Float16* __restrict__ wt,
                            int K, int Nn) {
    __shared__ float t[32][33];
    const int tid = threadIdx.x;
    const int n0 = blockIdx.x * 32, k0 = blockIdx.y * 32;
#pragma unroll
    for (int i = 0; i < 4; ++i) {
        const int idx = tid + i * 256;
        const int k = idx >> 5, n = idx & 31;
        t[k][n] = w[(size_t)(k0 + k) * Nn + n0 + n];
    }
    __syncthreads();
#pragma unroll
    for (int i = 0; i < 4; ++i) {
        const int idx = tid + i * 256;
        const int n = idx >> 5, k = idx & 31;
        wt[(size_t)(n0 + n) * K + k0 + k] = (_Float16)t[k][n];
    }
}

// ---------------------------------------------------------------------------
// Kernel 1: LayerNorm, fp32 -> f16  (one block per row)
// ---------------------------------------------------------------------------
__global__ void ln_kernel(const float* __restrict__ x,
                          const float* __restrict__ gamma,
                          const float* __restrict__ beta,
                          _Float16* __restrict__ xn) {
    __shared__ float red[256];
    const int row = blockIdx.x;
    const int tid = threadIdx.x;
    const float* xr = x + (size_t)row * D_;

    float v[4]; float s = 0.f;
#pragma unroll
    for (int i = 0; i < 4; ++i) { v[i] = xr[tid + i * 256]; s += v[i]; }
    red[tid] = s; __syncthreads();
    for (int o = 128; o > 0; o >>= 1) { if (tid < o) red[tid] += red[tid + o]; __syncthreads(); }
    const float mu = red[0] * (1.f / D_);
    __syncthreads();

    float sq = 0.f;
#pragma unroll
    for (int i = 0; i < 4; ++i) { const float d = v[i] - mu; sq += d * d; }
    red[tid] = sq; __syncthreads();
    for (int o = 128; o > 0; o >>= 1) { if (tid < o) red[tid] += red[tid + o]; __syncthreads(); }
    const float rstd = rsqrtf(red[0] * (1.f / D_) + 1e-5f);

    _Float16* xo = xn + (size_t)row * D_;
#pragma unroll
    for (int i = 0; i < 4; ++i) {
        const int c = tid + i * 256;
        xo[c] = (_Float16)((v[i] - mu) * rstd * gamma[c] + beta[c]);
    }
}

// ---------------------------------------------------------------------------
// Kernel 1b: per-head V transpose: Vb[bh][n][d] f16 -> VbT[bh][d][n] f16.
// ---------------------------------------------------------------------------
__global__ void vtr_kernel(const _Float16* __restrict__ Vb, _Float16* __restrict__ VbT) {
    __shared__ _Float16 t[32][33];
    const int tiles_n = N_ / 32, tiles_d = DH_ / 32;   // 64, 2
    const int per_bh = tiles_n * tiles_d;              // 128
    const int bh = blockIdx.x / per_bh;
    const int tile = blockIdx.x % per_bh;
    const int n0 = (tile % tiles_n) * 32, d0 = (tile / tiles_n) * 32;
    const _Float16* src = Vb + (size_t)bh * N_ * DH_;
    _Float16* dst = VbT + (size_t)bh * N_ * DH_;
    const int tid = threadIdx.x;
#pragma unroll
    for (int i = 0; i < 4; ++i) {
        const int idx = tid + i * 256;
        const int n = idx >> 5, d = idx & 31;
        t[n][d] = src[(size_t)(n0 + n) * DH_ + d0 + d];
    }
    __syncthreads();
#pragma unroll
    for (int i = 0; i < 4; ++i) {
        const int idx = tid + i * 256;
        const int d = idx >> 5, n = idx & 31;
        dst[(size_t)(d0 + d) * N_ + n0 + n] = t[n][d];
    }
}

// ---------------------------------------------------------------------------
// Async-stage a 128x32 f16 tile (rows stride 1024 f16 in memory) into padded
// LDS (stride LSTR f16). 512 16B chunks, 2 per thread.
// ---------------------------------------------------------------------------
__device__ __forceinline__ void stage_tile_async(const _Float16* __restrict__ src,
                                                 unsigned dst_off, int k0) {
    const int tid = threadIdx.x;
    const unsigned long long base = (unsigned long long)src + (unsigned long long)(k0 * 2);
#pragma unroll
    for (int i = 0; i < 2; ++i) {
        const int idx = tid + i * 256;
        const int r = idx >> 2, c16 = (idx & 3) * 16;
        async_b128(dst_off + (unsigned)(r * (LSTR * 2) + c16),
                   base + (unsigned long long)r * 2048ull + (unsigned)c16);
    }
}

// ---------------------------------------------------------------------------
// Shared WMMA GEMM mainloop: C[128x128] block tile, 8 waves each 32(M)x64(N).
// A f16 [*,1024] row-major, Bt f16 [Nn,1024] row-major (pre-transposed).
// Double-buffered async-to-LDS staging: DMA of tile k+1 overlaps WMMA on k.
// ---------------------------------------------------------------------------
__device__ __forceinline__ void gemm_mainloop(const _Float16* __restrict__ A,
                                              const _Float16* __restrict__ Bt,
                                              int row0, int col0,
                                              _Float16* __restrict__ la,   // [2][128][LSTR]
                                              _Float16* __restrict__ lbt,  // [2][128][LSTR]
                                              v8f acc[2][4]) {
    const int tid   = threadIdx.x;
    const int lane  = tid & 31;
    const int wave  = tid >> 5;
    const int wm    = wave & 3, wn = wave >> 2;
    const int khalf = lane >> 4, l16 = lane & 15;
    const unsigned la_off  = lds_off(la);
    const unsigned lbt_off = lds_off(lbt);
    const unsigned bufB = 128 * LSTR * 2;              // bytes per buffer
    const _Float16* Ab  = A  + (size_t)row0 * 1024;
    const _Float16* Btb = Bt + (size_t)col0 * 1024;

    stage_tile_async(Ab,  la_off,  0);
    stage_tile_async(Btb, lbt_off, 0);

    for (int k0 = 0; k0 < 1024; k0 += 32) {
        const int cur = (k0 >> 5) & 1;
        wait_async0();
        __syncthreads();
        if (k0 + 32 < 1024) {
            const unsigned nb = (unsigned)(cur ^ 1) * bufB;
            stage_tile_async(Ab,  la_off  + nb, k0 + 32);
            stage_tile_async(Btb, lbt_off + nb, k0 + 32);
        }
        const _Float16* lac = la  + cur * 128 * LSTR;
        const _Float16* lbc = lbt + cur * 128 * LSTR;

        Frag16 fa[2];
#pragma unroll
        for (int i = 0; i < 2; ++i)
#pragma unroll
            for (int v = 0; v < 8; ++v)
                fa[i].u[v] = *(const unsigned*)(lac + (wm * 32 + i * 16 + l16) * LSTR +
                                                kpair(v, khalf));
#pragma unroll
        for (int j = 0; j < 4; ++j) {
            Frag16 fb;
            const int nbase = wn * 64 + j * 16;
#pragma unroll
            for (int v = 0; v < 8; ++v)
                fb.u[v] = *(const unsigned*)(lbc + (nbase + l16) * LSTR + kpair(v, khalf));
#pragma unroll
            for (int i = 0; i < 2; ++i)
                acc[i][j] = wmma_f16(fa[i], fb, acc[i][j]);
        }
    }
}

// ---------------------------------------------------------------------------
// Kernel 2: QKV projection, scatter to [b,h,n,dh] f16 (Q pre-scaled by 1/8).
// ---------------------------------------------------------------------------
__global__ void qkv_kernel(const _Float16* __restrict__ xn,
                           const _Float16* __restrict__ wqkvT,
                           _Float16* __restrict__ Qb,
                           _Float16* __restrict__ Kb,
                           _Float16* __restrict__ Vb) {
    __shared__ _Float16 la[2 * 128 * LSTR];
    __shared__ _Float16 lbt[2 * 128 * LSTR];
    const int nblk = NQKV / 128;                      // 24
    const int row0 = (blockIdx.x / nblk) * 128;
    const int col0 = (blockIdx.x % nblk) * 128;
    v8f acc[2][4];
#pragma unroll
    for (int i = 0; i < 2; ++i)
#pragma unroll
        for (int j = 0; j < 4; ++j) acc[i][j] = {};
    gemm_mainloop(xn, wqkvT, row0, col0, la, lbt, acc);

    const int tid = threadIdx.x, lane = tid & 31, wave = tid >> 5;
    const int wm = wave & 3, wn = wave >> 2;
    const int khalf = lane >> 4, l16 = lane & 15;
#pragma unroll
    for (int j = 0; j < 4; ++j) {
        const int col   = col0 + wn * 64 + j * 16 + l16;
        const int which = col >> 10;                  // 0=Q 1=K 2=V (uniform per tile)
        const int rem   = col & 1023;
        const int h = rem >> 6, dh = rem & 63;
        _Float16* dst = (which == 0) ? Qb : ((which == 1) ? Kb : Vb);
        const float mul = (which == 0) ? 0.125f : 1.f;  // DH^-0.5
#pragma unroll
        for (int i = 0; i < 2; ++i)
#pragma unroll
            for (int r = 0; r < 8; ++r) {
                const int row = row0 + wm * 32 + i * 16 + r + 8 * khalf;
                const int b = row >> 11, n = row & 2047;
                dst[(((size_t)(b * H_ + h)) * N_ + n) * DH_ + dh] =
                    (_Float16)(acc[i][j][r] * mul);
            }
    }
}

// ---------------------------------------------------------------------------
// Kernel 3: flash attention. Block = 8 waves = 128 query rows of one (b,h).
// K and transposed-V tiles double-buffered via async-to-LDS; 32 keys/iter.
// ---------------------------------------------------------------------------
__global__ void attn_kernel(const _Float16* __restrict__ Qb,
                            const _Float16* __restrict__ Kb,
                            const _Float16* __restrict__ VbT,
                            const float* __restrict__ alibi,
                            const unsigned char* __restrict__ mask,
                            _Float16* __restrict__ ao) {
    __shared__ _Float16 kt[2 * 32 * KSTR];  // K tiles [key][d], padded
    __shared__ _Float16 vt[2 * 64 * LSTR];  // V^T tiles [d][key], padded
    __shared__ _Float16 pb[8][16 * 34];     // per-wave P staging [q][key]

    const int tid = threadIdx.x, lane = tid & 31, wave = tid >> 5;
    const int khalf = lane >> 4, l16 = lane & 15;
    const int qblocks = N_ / 128;           // 16
    const int bh = blockIdx.x / qblocks;
    const int qblk = blockIdx.x % qblocks;
    const int b = bh / H_, h = bh % H_;
    const int qbase = qblk * 128 + wave * 16;
    const unsigned kt_off = lds_off(kt);
    const unsigned vt_off = lds_off(vt);
    const unsigned ktB = 32 * KSTR * 2;     // bytes per K buffer
    const unsigned vtB = 64 * LSTR * 2;     // bytes per V buffer

    const _Float16* Qh  = Qb  + (size_t)bh * N_ * DH_;
    const unsigned long long KhA  = (unsigned long long)(Kb  + (size_t)bh * N_ * DH_);
    const unsigned long long VhTA = (unsigned long long)(VbT + (size_t)bh * N_ * DH_);
    const float*    ab = alibi + (size_t)h * N_ * N_;

    // Q fragments for both DH halves (constant over the key loop).
    Frag16 fq0, fq1;
#pragma unroll
    for (int v = 0; v < 8; ++v) {
        const int kp = kpair(v, khalf);
        fq0.u[v] = *(const unsigned*)(Qh + (size_t)(qbase + l16) * DH_ + kp);
        fq1.u[v] = *(const unsigned*)(Qh + (size_t)(qbase + l16) * DH_ + 32 + kp);
    }

    bool qm[8];
#pragma unroll
    for (int r = 0; r < 8; ++r) qm[r] = mask[b * N_ + qbase + r + 8 * khalf] != 0;

    float mrow[8], lrow[8];
    v8f o[4];
#pragma unroll
    for (int j = 0; j < 4; ++j) o[j] = {};
#pragma unroll
    for (int r = 0; r < 8; ++r) { mrow[r] = -1e30f; lrow[r] = 0.f; }

    // Stage K/V for one 32-key tile into buffer `buf` (2 async chunks/thread).
    const int krow = tid >> 3, kc16 = (tid & 7) * 16;   // K: 32 rows x 128B
    const int vrow = tid >> 2, vc16 = (tid & 3) * 16;   // V^T: 64 rows x 64B
    auto stage_kv = [&](int kb, int buf) {
        async_b128(kt_off + (unsigned)buf * ktB + (unsigned)(krow * (KSTR * 2) + kc16),
                   KhA + (unsigned long long)(kb + krow) * (DH_ * 2) + (unsigned)kc16);
        async_b128(vt_off + (unsigned)buf * vtB + (unsigned)(vrow * (LSTR * 2) + vc16),
                   VhTA + (unsigned long long)vrow * (N_ * 2) + (unsigned)(kb * 2 + vc16));
    };

    stage_kv(0, 0);

    for (int kb = 0; kb < N_; kb += 32) {
        const int cur = (kb >> 5) & 1;
        wait_async0();
        __syncthreads();
        if (kb + 32 < N_) stage_kv(kb + 32, cur ^ 1);
        const _Float16* ktc = kt + cur * 32 * KSTR;
        const _Float16* vtc = vt + cur * 64 * LSTR;

        // S = Q*K^T (+alibi, masks) for two 16-key subtiles.
        float s[2][8];
#pragma unroll
        for (int t = 0; t < 2; ++t) {
            Frag16 fb0, fb1;
#pragma unroll
            for (int v = 0; v < 8; ++v) {
                const int kp = kpair(v, khalf);
                fb0.u[v] = *(const unsigned*)(ktc + (t * 16 + l16) * KSTR + kp);
                fb1.u[v] = *(const unsigned*)(ktc + (t * 16 + l16) * KSTR + 32 + kp);
            }
            v8f sc = {};
            sc = wmma_f16(fq0, fb0, sc);
            sc = wmma_f16(fq1, fb1, sc);
            const int kcol = kb + t * 16 + l16;
            const bool km = mask[b * N_ + kcol] != 0;
#pragma unroll
            for (int r = 0; r < 8; ++r) {
                const int q = qbase + r + 8 * khalf;
                float val = sc[r] + ab[(size_t)q * N_ + kcol];
                if (!km)    val = NEGV;
                if (!qm[r]) val = NEGV;
                s[t][r] = val;
            }
        }

        // Online softmax (row reductions across the 16-lane half-wave).
        float alpha[8];
#pragma unroll
        for (int r = 0; r < 8; ++r) {
            float rmax = fmaxf(s[0][r], s[1][r]);
            for (int off = 8; off; off >>= 1)
                rmax = fmaxf(rmax, __shfl_xor(rmax, off, 16));
            const float mnew = fmaxf(mrow[r], rmax);
            const float a  = __expf(mrow[r] - mnew);
            const float p0 = __expf(s[0][r] - mnew);
            const float p1 = __expf(s[1][r] - mnew);
            float psum = p0 + p1;
            for (int off = 8; off; off >>= 1)
                psum += __shfl_xor(psum, off, 16);
            lrow[r] = lrow[r] * a + psum;
            mrow[r] = mnew;
            alpha[r] = a;
            s[0][r] = p0; s[1][r] = p1;
        }
#pragma unroll
        for (int j = 0; j < 4; ++j)
#pragma unroll
            for (int r = 0; r < 8; ++r) o[j][r] *= alpha[r];

        // C-layout -> A-layout for P via per-wave LDS round trip.
        _Float16* pw = pb[wave];
#pragma unroll
        for (int t = 0; t < 2; ++t)
#pragma unroll
            for (int r = 0; r < 8; ++r)
                pw[(r + 8 * khalf) * 34 + t * 16 + l16] = (_Float16)s[t][r];
        asm volatile("s_wait_dscnt 0" ::: "memory");   // wave-internal LDS RAW

        Frag16 pa;
#pragma unroll
        for (int v = 0; v < 8; ++v)
            pa.u[v] = *(const unsigned*)(pw + l16 * 34 + kpair(v, khalf));
#pragma unroll
        for (int j = 0; j < 4; ++j) {
            Frag16 fv;
#pragma unroll
            for (int v = 0; v < 8; ++v)
                fv.u[v] = *(const unsigned*)(vtc + (j * 16 + l16) * LSTR + kpair(v, khalf));
            o[j] = wmma_f16(pa, fv, o[j]);
        }
    }

    // Epilogue: normalize and write [b, n, h*DH] f16.
#pragma unroll
    for (int j = 0; j < 4; ++j) {
        const int d = j * 16 + l16;
#pragma unroll
        for (int r = 0; r < 8; ++r) {
            const int q = qbase + r + 8 * khalf;
            ao[((size_t)(b * N_ + q)) * (H_ * DH_) + h * DH_ + d] =
                (_Float16)(o[j][r] / lrow[r]);
        }
    }
}

// ---------------------------------------------------------------------------
// Kernel 4: output projection -> fp32 d_out.
// ---------------------------------------------------------------------------
__global__ void out_kernel(const _Float16* __restrict__ ao,
                           const _Float16* __restrict__ woutT,
                           float* __restrict__ out) {
    __shared__ _Float16 la[2 * 128 * LSTR];
    __shared__ _Float16 lbt[2 * 128 * LSTR];
    const int nblk = D_ / 128;                        // 8
    const int row0 = (blockIdx.x / nblk) * 128;
    const int col0 = (blockIdx.x % nblk) * 128;
    v8f acc[2][4];
#pragma unroll
    for (int i = 0; i < 2; ++i)
#pragma unroll
        for (int j = 0; j < 4; ++j) acc[i][j] = {};
    gemm_mainloop(ao, woutT, row0, col0, la, lbt, acc);

    const int tid = threadIdx.x, lane = tid & 31, wave = tid >> 5;
    const int wm = wave & 3, wn = wave >> 2;
    const int khalf = lane >> 4, l16 = lane & 15;
#pragma unroll
    for (int j = 0; j < 4; ++j) {
        const int col = col0 + wn * 64 + j * 16 + l16;
#pragma unroll
        for (int i = 0; i < 2; ++i)
#pragma unroll
            for (int r = 0; r < 8; ++r) {
                const int row = row0 + wm * 32 + i * 16 + r + 8 * khalf;
                out[(size_t)row * D_ + col] = acc[i][j][r];
            }
    }
}

// ---------------------------------------------------------------------------
extern "C" void kernel_launch(void* const* d_in, const int* in_sizes, int n_in,
                              void* d_out, int out_size, void* d_ws, size_t ws_size,
                              hipStream_t stream) {
    const float*         x     = (const float*)d_in[0];
    const float*         alibi = (const float*)d_in[1];
    const unsigned char* mask  = (const unsigned char*)d_in[2];
    const float*         gamma = (const float*)d_in[3];
    const float*         beta  = (const float*)d_in[4];
    const float*         wqkv  = (const float*)d_in[5];
    const float*         wout  = (const float*)d_in[6];
    float* out = (float*)d_out;

    char* ws = (char*)d_ws;                 // 56 MiB of f16 scratch
    _Float16* xn     = (_Float16*)(ws);                    //  8 MiB
    _Float16* Qb     = (_Float16*)(ws + (8ull  << 20));    //  8 MiB
    _Float16* Kb     = (_Float16*)(ws + (16ull << 20));    //  8 MiB
    _Float16* Vb     = (_Float16*)(ws + (24ull << 20));    //  8 MiB
    _Float16* ao     = (_Float16*)(ws + (32ull << 20));    //  8 MiB
    _Float16* wqkvT  = (_Float16*)(ws + (40ull << 20));    //  6 MiB
    _Float16* woutT  = (_Float16*)(ws + (46ull << 20));    //  2 MiB
    _Float16* VbT    = (_Float16*)(ws + (48ull << 20));    //  8 MiB

    wcvt_kernel<<<dim3(NQKV / 32, D_ / 32), 256, 0, stream>>>(wqkv, wqkvT, D_, NQKV);
    wcvt_kernel<<<dim3(D_ / 32,  D_ / 32),  256, 0, stream>>>(wout, woutT, D_, D_);
    ln_kernel  <<<B_ * N_, 256, 0, stream>>>(x, gamma, beta, xn);
    qkv_kernel <<<(B_ * N_ / 128) * (NQKV / 128), 256, 0, stream>>>(xn, wqkvT, Qb, Kb, Vb);
    vtr_kernel <<<B_ * H_ * (N_ / 32) * (DH_ / 32), 256, 0, stream>>>(Vb, VbT);
    attn_kernel<<<B_ * H_ * (N_ / 128), 256, 0, stream>>>(Qb, Kb, VbT, alibi, mask, ao);
    out_kernel <<<(B_ * N_ / 128) * (D_ / 128), 256, 0, stream>>>(ao, woutT, out);
}
